// SDARAttention_62801011802769
// MI455X (gfx1250) — matrix operand order repaired and checked by reading, not compiled, not run on hardware
//
#include <hip/hip_runtime.h>

// ---------------------------------------------------------------------------
// Problem constants (match reference)
// ---------------------------------------------------------------------------
constexpr int B_   = 2;
constexpr int S_   = 2048;
constexpr int HID  = 2048;
constexpr int NH   = 32;
constexpr int NKV  = 8;
constexpr int D    = 64;
constexpr int G    = NH / NKV;   // 4
constexpr int BS   = B_ * S_;    // 4096
constexpr float EPS       = 1e-6f;
constexpr float SCALE     = 0.125f;               // D^-0.5
constexpr float LOG_THETA = 9.210340371976184f;   // ln(10000)

// ---------------------------------------------------------------------------
// WMMA vector types (CDNA5 wave32: bf16 operands = 16 elems, f32 acc = 8)
// ---------------------------------------------------------------------------
typedef __attribute__((ext_vector_type(16))) __bf16 v16bf;
typedef __attribute__((ext_vector_type(8)))  __bf16 v8bf;
typedef __attribute__((ext_vector_type(8)))  float  v8f;
typedef __attribute__((ext_vector_type(4)))  int    v4i;

__device__ inline v16bf cat8(v8bf lo, v8bf hi) {
  return __builtin_shufflevector(lo, hi, 0,1,2,3,4,5,6,7,8,9,10,11,12,13,14,15);
}

__device__ inline v8f wmma_bf16(v16bf a, v16bf b, v8f c) {
  return __builtin_amdgcn_wmma_f32_16x16x32_bf16(false, a, false, b,
                                                 (short)0, c, false, false);
}

// ---------------------------------------------------------------------------
// CDNA5 async global->LDS staging (ASYNCcnt path), guarded so the file still
// compiles if this toolchain doesn't expose the builtins.
// Builtin signature (from hipcc diagnostic): param0 = AS1 (global) v4i*,
// so cast through address-space-qualified v4i pointee typedefs.
// ---------------------------------------------------------------------------
#if defined(__HIP_DEVICE_COMPILE__) &&                                        \
    __has_builtin(__builtin_amdgcn_global_load_async_to_lds_b128) &&          \
    __has_builtin(__builtin_amdgcn_s_wait_asynccnt)
#define USE_ASYNC_LDS 1
#else
#define USE_ASYNC_LDS 0
#endif

__device__ inline void stage_b128(const __bf16* gsrc, __bf16* ldst) {
#if USE_ASYNC_LDS
  typedef __attribute__((address_space(1))) v4i gv4i_t;
  typedef __attribute__((address_space(3))) v4i lv4i_t;
  __builtin_amdgcn_global_load_async_to_lds_b128(
      (gv4i_t*)gsrc, (lv4i_t*)ldst, 0, 0);
#else
  *(v8bf*)ldst = *(const v8bf*)gsrc;
#endif
}

__device__ inline void wait_stage() {
#if USE_ASYNC_LDS
  __builtin_amdgcn_s_wait_asynccnt(0);
#endif
}

// ---------------------------------------------------------------------------
// fp32 -> bf16 convert
// ---------------------------------------------------------------------------
__global__ void cvt_f32_bf16(const float* __restrict__ in,
                             __bf16* __restrict__ out, int n) {
  int i = blockIdx.x * blockDim.x + threadIdx.x;
  if (i < n) out[i] = (__bf16)in[i];
}

// ---------------------------------------------------------------------------
// Generic GEMM: C[M,N] (f32) = A[M,K] (bf16 row-major) * W[N,K]^T (bf16 row-major)
// Block = 256 threads = 8 waves; block tile 128(M) x 64(N); wave tile 16 x 64.
// B tile (64 rows x 32 k) is staged through LDS (async, double-buffered) and
// shared by all 8 waves; A fragments are per-wave direct global loads.
// A fragment (16x32): lane = (r, hf); elems 0-15 = K[hf*16 .. hf*16+15]
// B fragment (32x16): lane = (col r, hf); elems 0-15 = K[hf*16 .. hf*16+15]
// C fragment: VGPR i, lane(r,hf) -> (M = i + 8*hf, N = r)
// ---------------------------------------------------------------------------
__global__ __launch_bounds__(256)
void gemm_bf16(const __bf16* __restrict__ A, const __bf16* __restrict__ W,
               float* __restrict__ C, int M, int N, int K) {
  __shared__ __align__(32) __bf16 bsh[2][64][48];  // 48-elem stride: 32B-aligned frags

  const int tid  = threadIdx.x;
  const int lane = tid & 31;
  const int wv   = tid >> 5;
  const int r    = lane & 15;
  const int hf   = lane >> 4;
  const int m0   = blockIdx.y * 128 + wv * 16;
  const int n0   = blockIdx.x * 64;
  const long arow = (long)(m0 + r) * K;

  // Cooperative B staging: thread -> (row bn of 64, 8-elem k-chunk bk)
  const int bn = tid >> 2;
  const int bk = (tid & 3) * 8;
  const __bf16* wrow = W + (long)(n0 + bn) * K + bk;

  stage_b128(wrow, &bsh[0][bn][bk]);

  v8f acc[4] = {};
  for (int kk = 0; kk < K; kk += 32) {
    const int buf = (kk >> 5) & 1;
    wait_stage();       // this thread's stage for `buf` has landed in LDS
    __syncthreads();    // everyone's stage visible; prior reads of buf^1 done
    if (kk + 32 < K) stage_b128(wrow + kk + 32, &bsh[buf ^ 1][bn][bk]);

    v8bf alo = *(const v8bf*)(A + arow + kk + hf * 16);
    v8bf ahi = *(const v8bf*)(A + arow + kk + hf * 16 + 8);
    v16bf af = cat8(alo, ahi);
#pragma unroll
    for (int j = 0; j < 4; ++j) {
      v16bf bfrag = *(const v16bf*)(&bsh[buf][j * 16 + r][hf * 16]);
      acc[j] = wmma_bf16(af, bfrag, acc[j]);
    }
  }
#pragma unroll
  for (int j = 0; j < 4; ++j)
#pragma unroll
    for (int i = 0; i < 8; ++i)
      C[(long)(m0 + i + 8 * hf) * N + n0 + j * 16 + r] = acc[j][i];
}

// ---------------------------------------------------------------------------
// Per-head RMSNorm + RoPE (+optional score scale folded into Q).
// in : [B, S, H, D] f32 ;  out : [B, H, S, D] bf16 (head-major for attention)
// One wave per (b,s,h); lane covers d=lane and d=lane+32 (rotate_half pair).
// ---------------------------------------------------------------------------
__global__ __launch_bounds__(256)
void norm_rope(const float* __restrict__ in, const int* __restrict__ pos,
               const float* __restrict__ w, __bf16* __restrict__ out,
               int H, float scale) {
  const int lane = threadIdx.x & 31;
  const int inst = (blockIdx.x * blockDim.x + threadIdx.x) >> 5;
  const int h = inst % H;
  const int s = (inst / H) % S_;
  const int b = inst / (H * S_);

  const float* px = in + ((long)((b * S_ + s) * H + h)) * D;
  float x0 = px[lane];
  float x1 = px[lane + 32];
  float ss = x0 * x0 + x1 * x1;
#pragma unroll
  for (int m = 16; m >= 1; m >>= 1) ss += __shfl_xor(ss, m, 32);
  float rinv = rsqrtf(ss * (1.0f / D) + EPS);
  float y0 = x0 * rinv * w[lane];
  float y1 = x1 * rinv * w[lane + 32];

  float p    = (float)pos[b * S_ + s];
  float freq = __expf(-((float)(2 * lane) / (float)D) * LOG_THETA);
  float ang  = p * freq;
  float c = cosf(ang), sn = sinf(ang);
  float o0 = (y0 * c - y1 * sn) * scale;
  float o1 = (y1 * c + y0 * sn) * scale;

  __bf16* po = out + ((long)((b * H + h) * S_ + s)) * D;
  po[lane]      = (__bf16)o0;
  po[lane + 32] = (__bf16)o1;
}

// ---------------------------------------------------------------------------
// V: [B, S, NKV, D] f32  ->  Vt: [B, NKV, D, S] bf16 (so PV B-operand loads
// are contiguous rows, same "W pattern" as the GEMMs).
// ---------------------------------------------------------------------------
__global__ void v_transpose(const float* __restrict__ in,
                            __bf16* __restrict__ out, int total) {
  int i = blockIdx.x * blockDim.x + threadIdx.x;
  if (i >= total) return;
  int d  = i % D;
  int hk = (i / D) % NKV;
  int s  = (i / (D * NKV)) % S_;
  int b  = i / (D * NKV * S_);
  out[(((long)(b * NKV + hk) * D + d) * S_) + s] = (__bf16)in[i];
}

// ---------------------------------------------------------------------------
// Fused flash attention (non-causal, full softmax over S, GQA).
// Q : [B, NH, S, D] bf16 (already RMSNormed, RoPEd, *SCALE)
// K : [B, NKV, S, D] bf16 (RMSNormed, RoPEd)
// Vt: [B, NKV, D, S] bf16
// O : [B, S, NH*D] bf16 (token-major, feeds final GEMM)
// Block = 256 thr = 8 waves; wave owns 16 query rows; key loop in chunks of 32.
// ---------------------------------------------------------------------------
__global__ __launch_bounds__(256)
void flash_attn(const __bf16* __restrict__ Q, const __bf16* __restrict__ Kc,
                const __bf16* __restrict__ Vt, __bf16* __restrict__ O) {
  __shared__ __align__(32) __bf16 lds[8][16][32];   // per-wave P tile

  const int lane = threadIdx.x & 31;
  const int wv   = threadIdx.x >> 5;
  const int r    = lane & 15;
  const int hf   = lane >> 4;
  const int h  = blockIdx.y;
  const int b  = blockIdx.z;
  const int hk = h / G;
  const int q0 = blockIdx.x * 128 + wv * 16;

  // Q fragments for d-chunks {0..31}, {32..63}
  const __bf16* qb = Q + (((long)(b * NH + h) * S_) + q0 + r) * D;
  v16bf aq[2];
#pragma unroll
  for (int c2 = 0; c2 < 2; ++c2) {
    v8bf lo = *(const v8bf*)(qb + c2 * 32 + hf * 16);
    v8bf hi = *(const v8bf*)(qb + c2 * 32 + hf * 16 + 8);
    aq[c2] = cat8(lo, hi);
  }

  const __bf16* kb = Kc + ((long)(b * NKV + hk) * S_) * D;
  const __bf16* vb = Vt + ((long)(b * NKV + hk) * D) * S_;

  v8f acc[4] = {};
  float mrow[8], lrow[8];
#pragma unroll
  for (int i = 0; i < 8; ++i) { mrow[i] = -1e30f; lrow[i] = 0.f; }

  for (int sk = 0; sk < S_; sk += 32) {
    if (sk + 32 < S_) {   // prefetch next K/V tiles (global_prefetch_b8)
      __builtin_prefetch(kb + (long)(sk + 32 + r) * D, 0, 1);
      __builtin_prefetch(vb + (long)r * S_ + sk + 32, 0, 1);
    }
    // ---- scores: 16q x 32k = two 16x16 C frags, 2 WMMAs each (D=64) ----
    v8f sc[2];
#pragma unroll
    for (int j = 0; j < 2; ++j) {
      const __bf16* kp = kb + (long)(sk + j * 16 + r) * D + hf * 16;
      v16bf b0 = *(const v16bf*)(kp);        // d = 0..31 (across halves)
      v16bf b1 = *(const v16bf*)(kp + 32);   // d = 32..63
      v8f z = {};
      z     = wmma_bf16(aq[0], b0, z);
      sc[j] = wmma_bf16(aq[1], b1, z);
    }
    // ---- online softmax over the 32-key chunk ----
#pragma unroll
    for (int i = 0; i < 8; ++i) {
      float mx = fmaxf(sc[0][i], sc[1][i]);
#pragma unroll
      for (int m = 8; m >= 1; m >>= 1) mx = fmaxf(mx, __shfl_xor(mx, m, 32));
      float mnew = fmaxf(mrow[i], mx);
      float corr = __expf(mrow[i] - mnew);
      float p0 = __expf(sc[0][i] - mnew);
      float p1 = __expf(sc[1][i] - mnew);
      float rs = p0 + p1;
#pragma unroll
      for (int m = 8; m >= 1; m >>= 1) rs += __shfl_xor(rs, m, 32);
      lrow[i] = lrow[i] * corr + rs;
      mrow[i] = mnew;
#pragma unroll
      for (int db = 0; db < 4; ++db) acc[db][i] *= corr;
      // stage P (C-layout) into LDS for A-layout reread
      lds[wv][i + 8 * hf][r]      = (__bf16)p0;
      lds[wv][i + 8 * hf][16 + r] = (__bf16)p1;
    }
    // ---- reread P as 16x32 A fragment (wave-local, DS ops are in-order) ----
    v8bf plo = *(const v8bf*)(&lds[wv][r][hf * 16]);
    v8bf phi = *(const v8bf*)(&lds[wv][r][hf * 16 + 8]);
    v16bf pf = cat8(plo, phi);
    // ---- acc += P * V  (4 WMMAs cover d = 0..63) ----
#pragma unroll
    for (int db = 0; db < 4; ++db) {
      const __bf16* vp = vb + (long)(db * 16 + r) * S_ + sk + hf * 16;
      v16bf bv = *(const v16bf*)vp;
      acc[db] = wmma_bf16(pf, bv, acc[db]);
    }
  }
  // ---- epilogue: normalize and write token-major bf16 ----
#pragma unroll
  for (int i = 0; i < 8; ++i) {
    float inv = 1.0f / lrow[i];
    int row = q0 + i + 8 * hf;
    __bf16* po = O + ((long)(b * S_ + row)) * (NH * D) + h * D;
#pragma unroll
    for (int db = 0; db < 4; ++db)
      po[db * 16 + r] = (__bf16)(acc[db][i] * inv);
  }
}

// ---------------------------------------------------------------------------
// Host-side orchestration
// ---------------------------------------------------------------------------
extern "C" void kernel_launch(void* const* d_in, const int* in_sizes, int n_in,
                              void* d_out, int out_size, void* d_ws, size_t ws_size,
                              hipStream_t stream) {
  const int*   positions = (const int*)d_in[0];
  const float* hs = (const float*)d_in[1];
  const float* Wq = (const float*)d_in[2];
  const float* Wk = (const float*)d_in[3];
  const float* Wv = (const float*)d_in[4];
  const float* Wo = (const float*)d_in[5];
  const float* qn = (const float*)d_in[6];
  const float* kn = (const float*)d_in[7];
  float* outp = (float*)d_out;

  char* ws = (char*)d_ws;
  size_t off = 0;
  auto alloc = [&](size_t bytes) -> void* {
    void* p = ws + off;
    off += (bytes + 255) & ~(size_t)255;
    return p;
  };

  __bf16* Xb  = (__bf16*)alloc((size_t)BS * HID * 2);
  __bf16* Wqb = (__bf16*)alloc((size_t)NH * D * HID * 2);
  __bf16* Wkb = (__bf16*)alloc((size_t)NKV * D * HID * 2);
  __bf16* Wvb = (__bf16*)alloc((size_t)NKV * D * HID * 2);
  __bf16* Wob = (__bf16*)alloc((size_t)HID * NH * D * 2);
  float*  Qf  = (float*)alloc((size_t)BS * NH * D * 4);
  float*  Kf  = (float*)alloc((size_t)BS * NKV * D * 4);
  float*  Vf  = (float*)alloc((size_t)BS * NKV * D * 4);
  __bf16* Qb  = (__bf16*)alloc((size_t)BS * NH * D * 2);
  __bf16* Kb  = (__bf16*)alloc((size_t)BS * NKV * D * 2);
  __bf16* Vtb = (__bf16*)alloc((size_t)BS * NKV * D * 2);
  __bf16* Ob  = (__bf16*)alloc((size_t)BS * NH * D * 2);

  auto cvt = [&](const float* src, __bf16* dst, int n) {
    cvt_f32_bf16<<<(n + 255) / 256, 256, 0, stream>>>(src, dst, n);
  };
  cvt(hs, Xb, BS * HID);
  cvt(Wq, Wqb, NH * D * HID);
  cvt(Wk, Wkb, NKV * D * HID);
  cvt(Wv, Wvb, NKV * D * HID);
  cvt(Wo, Wob, HID * NH * D);

  // Projections: X[BS,HID] @ W^T
  gemm_bf16<<<dim3((NH * D) / 64, BS / 128), 256, 0, stream>>>(Xb, Wqb, Qf, BS, NH * D, HID);
  gemm_bf16<<<dim3((NKV * D) / 64, BS / 128), 256, 0, stream>>>(Xb, Wkb, Kf, BS, NKV * D, HID);
  gemm_bf16<<<dim3((NKV * D) / 64, BS / 128), 256, 0, stream>>>(Xb, Wvb, Vf, BS, NKV * D, HID);

  // RMSNorm + RoPE (+SCALE folded into Q), head-major repack
  norm_rope<<<(B_ * S_ * NH) / 8, 256, 0, stream>>>(Qf, positions, qn, Qb, NH, SCALE);
  norm_rope<<<(B_ * S_ * NKV) / 8, 256, 0, stream>>>(Kf, positions, kn, Kb, NKV, 1.0f);

  // V -> [B,NKV,D,S] bf16
  v_transpose<<<(BS * NKV * D) / 256, 256, 0, stream>>>(Vf, Vtb, BS * NKV * D);

  // Fused attention
  flash_attn<<<dim3(S_ / 128, NH, B_), 256, 0, stream>>>(Qb, Kb, Vtb, Ob);

  // Output projection -> f32 result
  gemm_bf16<<<dim3(HID / 64, BS / 128), 256, 0, stream>>>(Ob, Wob, outp, BS, HID, NH * D);

  (void)in_sizes; (void)n_in; (void)out_size; (void)ws_size;
}
